// PointNet2SAModule_25434796327487
// MI455X (gfx1250) — compile-verified, matrix operand order checked by reading, not run on hardware
//
#include <hip/hip_runtime.h>
#include <hip/hip_bf16.h>

typedef _Float16 h16;
typedef __attribute__((ext_vector_type(16))) _Float16 v16h;
typedef __attribute__((ext_vector_type(8)))  _Float16 v8h;
typedef __attribute__((ext_vector_type(8)))  float    v8f;

#define B_CLOUDS 8
#define N_PER    4096
#define FDIM     64
#define S_SAMP   1024
#define KNBR     64
#define R2       0.04f
#define NSAMP    (B_CLOUDS * S_SAMP)   // 8192
#define NEGBIG   (-1.0e30f)

// ---------------------------------------------------------------------------
// Fragment loader: 16-bit A/B WMMA fragment (16 rows x 32 K) per ISA layout:
//   lanes 0-15 : row = row0+lane,    K = kc+0..7  (v0..3) and kc+16..23 (v4..7)
//   lanes 16-31: row = row0+lane-16, K = kc+8..15           kc+24..31
// => two contiguous 16-byte loads per lane.
// ---------------------------------------------------------------------------
__device__ __forceinline__ v16h load_frag(const h16* base, int pitch, int row0,
                                          int kc, int lane) {
  int r  = row0 + (lane & 15);
  int kb = kc + ((lane >> 4) << 3);
  const h16* p = base + r * pitch + kb;
  v8h a0 = *(const v8h*)(p);
  v8h a1 = *(const v8h*)(p + 16);
  v16h out;
#pragma unroll
  for (int i = 0; i < 8; ++i) { out[i] = a0[i]; out[i + 8] = a1[i]; }
  return out;
}

// One MLP layer for one sample handled by one wave32:
//   Aout[64][NOUT] = relu(Ain[64][KD] @ Wt^T + bias),  Wt is [NOUT][KD] f16.
template <int KD, int NOUT>
__device__ __forceinline__ void mlp_layer(const h16* Ain, const h16* Wt,
                                          const float* bias, h16* Aout,
                                          int lane) {
#pragma unroll 1
  for (int nt = 0; nt < NOUT / 16; ++nt) {
    float bv = bias[nt * 16 + (lane & 15)];
    v8f acc[4];
#pragma unroll
    for (int mt = 0; mt < 4; ++mt)
#pragma unroll
      for (int i = 0; i < 8; ++i) acc[mt][i] = bv;

#pragma unroll 1
    for (int kc = 0; kc < KD; kc += 32) {
      v16h bf = load_frag(Wt, KD, nt * 16, kc, lane);
#pragma unroll
      for (int mt = 0; mt < 4; ++mt) {
        v16h af = load_frag(Ain, KD, mt * 16, kc, lane);
        acc[mt] = __builtin_amdgcn_wmma_f32_16x16x32_f16(
            false, af, false, bf, (short)0, acc[mt], false, false);
      }
    }
    int col   = nt * 16 + (lane & 15);
    int rbase = (lane >> 4) << 3;
#pragma unroll
    for (int mt = 0; mt < 4; ++mt)
#pragma unroll
      for (int r = 0; r < 8; ++r) {
        float v = acc[mt][r];
        v = v > 0.f ? v : 0.f;
        Aout[(mt * 16 + rbase + r) * NOUT + col] = (h16)v;
      }
  }
}

// ---------------------------------------------------------------------------
// Kernel 0: weight prep — transpose to [NOUT][K] row-major f16 (W1 padded 67->96)
// ---------------------------------------------------------------------------
__global__ void prep_kernel(const float* W1, const float* W2, const float* W3,
                            h16* W1t, h16* W2t, h16* W3t) {
  int i = blockIdx.x * blockDim.x + threadIdx.x;
  if (i < 128 * 96) {
    int n = i / 96, k = i % 96;
    W1t[i] = (h16)(k < 67 ? W1[k * 128 + n] : 0.f);
  }
  if (i < 128 * 128) {
    int n = i / 128, k = i % 128;
    W2t[i] = (h16)W2[k * 128 + n];
  }
  if (i < 256 * 128) {
    int n = i / 128, k = i % 128;
    W3t[i] = (h16)W3[k * 256 + n];
  }
}

// ---------------------------------------------------------------------------
// Kernel 1: farthest point sampling, one block (1024 thr) per cloud.
// Key packing: ((bits(min_d))<<32) | (4095-n) -> max-reduce picks largest
// distance, ties resolved to the LOWEST index (matches jnp.argmax).
// ---------------------------------------------------------------------------
__global__ void fps_kernel(const float* pos, int* idx_s) {
  int cloud = blockIdx.x;
  int tid   = threadIdx.x;  // 0..1023
  const float* P = pos + (size_t)cloud * N_PER * 3;

  float px[4], py[4], pz[4], md[4];
  float sx = P[0], sy = P[1], sz = P[2];
#pragma unroll
  for (int t = 0; t < 4; ++t) {
    int n = tid + t * 1024;
    px[t] = P[n * 3 + 0]; py[t] = P[n * 3 + 1]; pz[t] = P[n * 3 + 2];
    float dx = px[t] - sx, dy = py[t] - sy, dz = pz[t] - sz;
    md[t] = dx * dx + dy * dy + dz * dz;
  }
  if (tid == 0) idx_s[cloud * S_SAMP] = 0;

  __shared__ unsigned long long sk[1024];
  __shared__ float bc[3];

  for (int s = 1; s < S_SAMP; ++s) {
    unsigned long long best = 0ull;
#pragma unroll
    for (int t = 0; t < 4; ++t) {
      int n = tid + t * 1024;
      unsigned long long key =
          ((unsigned long long)__float_as_uint(md[t]) << 32) |
          (unsigned)(4095 - n);
      if (key > best) best = key;
    }
    sk[tid] = best;
    __syncthreads();
    for (int off = 512; off > 0; off >>= 1) {
      if (tid < off && sk[tid + off] > sk[tid]) sk[tid] = sk[tid + off];
      __syncthreads();
    }
    unsigned long long win = sk[0];
    int n = 4095 - (int)(win & 0xFFFFFFFFull);
    if ((n & 1023) == tid) {
      int t = n >> 10;
      bc[0] = px[t]; bc[1] = py[t]; bc[2] = pz[t];
    }
    if (tid == 0) idx_s[cloud * S_SAMP + s] = n;
    __syncthreads();
    float bx = bc[0], by = bc[1], bz = bc[2];
#pragma unroll
    for (int t = 0; t < 4; ++t) {
      float dx = px[t] - bx, dy = py[t] - by, dz = pz[t] - bz;
      float d = dx * dx + dy * dy + dz * dz;
      md[t] = fminf(md[t], d);
    }
    __syncthreads();
  }
}

// ---------------------------------------------------------------------------
// Kernel 2: K=64 nearest neighbors within radius, one block (256 thr)/sample.
// 64 rounds of block argmin on ((bits(d2))<<32)|n keys (tie -> lowest index).
// Stores global neighbor row indices + 64-bit validity mask (d2 <= R^2).
// ---------------------------------------------------------------------------
__global__ void knn_kernel(const float* pos, const int* idx_s, int* nbr,
                           unsigned* valid) {
  int samp  = blockIdx.x;      // 0..8191
  int cloud = samp >> 10;
  int tid   = threadIdx.x;     // 0..255
  const float* P = pos + (size_t)cloud * N_PER * 3;
  int sidx = idx_s[samp];
  float sx = P[sidx * 3 + 0], sy = P[sidx * 3 + 1], sz = P[sidx * 3 + 2];

  float d[16];
#pragma unroll
  for (int t = 0; t < 16; ++t) {
    int n = tid + t * 256;
    float dx = P[n * 3 + 0] - sx, dy = P[n * 3 + 1] - sy, dz = P[n * 3 + 2] - sz;
    d[t] = dx * dx + dy * dy + dz * dz;
  }

  __shared__ unsigned long long sk[256];
  unsigned vm0 = 0, vm1 = 0;

  for (int k = 0; k < KNBR; ++k) {
    unsigned long long best = 0xFFFFFFFFFFFFFFFFull;
#pragma unroll
    for (int t = 0; t < 16; ++t) {
      int n = tid + t * 256;
      unsigned long long key =
          ((unsigned long long)__float_as_uint(d[t]) << 32) | (unsigned)n;
      if (key < best) best = key;
    }
    sk[tid] = best;
    __syncthreads();
    for (int off = 128; off > 0; off >>= 1) {
      if (tid < off && sk[tid + off] < sk[tid]) sk[tid] = sk[tid + off];
      __syncthreads();
    }
    unsigned long long win = sk[0];
    int   n  = (int)(win & 0xFFFFFFFFull);
    float dw = __uint_as_float((unsigned)(win >> 32));
    if ((n & 255) == tid) d[n >> 8] = 3.402823466e38f;  // exclude
    if (tid == 0) {
      nbr[samp * KNBR + k] = cloud * N_PER + n;
      if (dw <= R2) { if (k < 32) vm0 |= 1u << k; else vm1 |= 1u << (k - 32); }
    }
    __syncthreads();
  }
  if (tid == 0) { valid[samp * 2] = vm0; valid[samp * 2 + 1] = vm1; }
}

// ---------------------------------------------------------------------------
// Kernel 3: edge MLP (WMMA f16 -> f32) + masked max aggregation.
// One wave32 per sample; H-matrices staged in LDS, weights from L2.
// ---------------------------------------------------------------------------
__global__ __launch_bounds__(32) void mlp_kernel(
    const float* x, const float* pos, const int* idx_s, const int* nbr,
    const unsigned* valid, const h16* W1t, const float* b1, const h16* W2t,
    const float* b2, const h16* W3t, const float* b3, float* out) {
  __shared__ __align__(16) h16 bufA[64 * 128];  // H1 (64x96) then H3 (64x128)
  __shared__ __align__(16) h16 bufB[64 * 128];  // H2 (64x128)

  int samp  = blockIdx.x;
  int lane  = threadIdx.x;
  int cloud = samp >> 10;
  int sidx  = idx_s[samp];
  const float* pS = pos + (size_t)(cloud * N_PER + sidx) * 3;
  float sx = pS[0], sy = pS[1], sz = pS[2];

  // Build H1 = [x_j | rel | 0-pad] as f16, pitch 96.
  for (int j = lane; j < KNBR; j += 32) {
    int g = nbr[samp * KNBR + j];
    const float* xr = x + (size_t)g * FDIM;
    h16* row = bufA + j * 96;
#pragma unroll 8
    for (int c = 0; c < FDIM; ++c) row[c] = (h16)xr[c];
    const float* pj = pos + (size_t)g * 3;
    row[64] = (h16)(pj[0] - sx);
    row[65] = (h16)(pj[1] - sy);
    row[66] = (h16)(pj[2] - sz);
#pragma unroll
    for (int c = 67; c < 96; ++c) row[c] = (h16)0.f;
  }
  __syncthreads();

  mlp_layer<96, 128>(bufA, W1t, b1, bufB, lane);   // H2 = relu(H1 W1 + b1)
  __syncthreads();
  mlp_layer<128, 128>(bufB, W2t, b2, bufA, lane);  // H3 = relu(H2 W2 + b2)
  __syncthreads();

  // Layer 3 + masked max over the 64 neighbor rows, straight from acc VGPRs.
  unsigned v0 = valid[samp * 2], v1 = valid[samp * 2 + 1];
#pragma unroll 1
  for (int nt = 0; nt < 16; ++nt) {
    float bv = b3[nt * 16 + (lane & 15)];
    v8f acc[4];
#pragma unroll
    for (int mt = 0; mt < 4; ++mt)
#pragma unroll
      for (int i = 0; i < 8; ++i) acc[mt][i] = bv;
#pragma unroll 1
    for (int kc = 0; kc < 128; kc += 32) {
      v16h bf = load_frag(W3t, 128, nt * 16, kc, lane);
#pragma unroll
      for (int mt = 0; mt < 4; ++mt) {
        v16h af = load_frag(bufA, 128, mt * 16, kc, lane);
        acc[mt] = __builtin_amdgcn_wmma_f32_16x16x32_f16(
            false, af, false, bf, (short)0, acc[mt], false, false);
      }
    }
    float mx = NEGBIG;
#pragma unroll
    for (int mt = 0; mt < 4; ++mt) {
      int r0 = mt * 16 + ((lane >> 4) << 3);
#pragma unroll
      for (int r = 0; r < 8; ++r) {
        int row = r0 + r;
        bool ok = (row < 32) ? ((v0 >> row) & 1u) : ((v1 >> (row - 32)) & 1u);
        float v = acc[mt][r];
        v = v > 0.f ? v : 0.f;
        if (ok) mx = fmaxf(mx, v);
      }
    }
    mx = fmaxf(mx, __shfl_xor(mx, 16, 32));  // combine row-halves (column-matched lanes)
    if (mx < -5.0e29f) mx = 0.f;
    if (lane < 16) out[(size_t)samp * 256 + nt * 16 + lane] = mx;
  }

  if (lane == 0) {  // pos1 / batch1 epilogue
    float* p1 = out + (size_t)NSAMP * 256 + (size_t)samp * 3;
    p1[0] = sx; p1[1] = sy; p1[2] = sz;
    out[(size_t)NSAMP * 259 + samp] = (float)cloud;
  }
}

// ---------------------------------------------------------------------------
extern "C" void kernel_launch(void* const* d_in, const int* in_sizes, int n_in,
                              void* d_out, int out_size, void* d_ws,
                              size_t ws_size, hipStream_t stream) {
  const float* x   = (const float*)d_in[0];
  const float* pos = (const float*)d_in[1];
  // d_in[2] = batch (int) — value is cloud id, reconstructed directly.
  const float* W1 = (const float*)d_in[3];
  const float* b1 = (const float*)d_in[4];
  const float* W2 = (const float*)d_in[5];
  const float* b2 = (const float*)d_in[6];
  const float* W3 = (const float*)d_in[7];
  const float* b3 = (const float*)d_in[8];
  float* out = (float*)d_out;

  char* ws = (char*)d_ws;
  int*      idx_s = (int*)(ws);                               // 8192 ints
  int*      nbr   = (int*)(ws + 32768);                       // 8192*64 ints
  unsigned* valid = (unsigned*)(ws + 32768 + 2097152);        // 8192*2
  h16*      W1t   = (h16*)(ws + 2195456);                     // 128*96
  h16*      W2t   = (h16*)(ws + 2220032);                     // 128*128
  h16*      W3t   = (h16*)(ws + 2252800);                     // 256*128

  prep_kernel<<<128, 256, 0, stream>>>(W1, W2, W3, W1t, W2t, W3t);
  fps_kernel<<<B_CLOUDS, 1024, 0, stream>>>(pos, idx_s);
  knn_kernel<<<NSAMP, 256, 0, stream>>>(pos, idx_s, nbr, valid);
  mlp_kernel<<<NSAMP, 32, 0, stream>>>(x, pos, idx_s, nbr, valid, W1t, b1, W2t,
                                       b2, W3t, b3, out);
}
